// GraphAttentionLayer_730144441122
// MI455X (gfx1250) — compile-verified, matrix operand order checked by reading, not run on hardware
//
#include <hip/hip_runtime.h>
#include <hip/hip_bf16.h>

// ---------------------------------------------------------------------------
// GAT layer on MI455X (gfx1250, wave32, WMMA).
//   N = 8192, IN_F = 512, OUT_F = 256, ALPHA = 0.2, NEG_INF = -9e15
// ---------------------------------------------------------------------------

#define GAT_N     8192
#define GAT_INF   512
#define GAT_OUTF  256
#define GAT_ALPHA 0.2f
#define GAT_NEGINF (-9.0e15f)

typedef _Float16 v16h __attribute__((ext_vector_type(16)));
typedef _Float16 v8h  __attribute__((ext_vector_type(8)));
typedef float    v8f  __attribute__((ext_vector_type(8)));

// ----------------------------- helpers -------------------------------------

__device__ __forceinline__ unsigned gat_ballot32(bool pred) {
#if __has_builtin(__builtin_amdgcn_ballot_w32)
    return __builtin_amdgcn_ballot_w32(pred);
#else
    return (unsigned)__ballot(pred);
#endif
}

// combine two (max, sumexp) softmax partials
__device__ __forceinline__ void sm_combine(float& m, float& s, float mo, float so) {
    float mn = fmaxf(m, mo);
    s = s * __expf(m - mn) + so * __expf(mo - mn);
    m = mn;
}

__device__ __forceinline__ v16h join16(v8h lo, v8h hi) {
    v16h r;
#pragma unroll
    for (int i = 0; i < 8; ++i) { r[i] = lo[i]; r[i + 8] = hi[i]; }
    return r;
}

// ----------------------- 1) precision conversion ---------------------------

__global__ __launch_bounds__(256) void k_cvt_input(const float* __restrict__ in,
                                                   _Float16* __restrict__ out, int n) {
    int i = blockIdx.x * 256 + threadIdx.x;
    if (i < n) out[i] = (_Float16)in[i];
}

// W[512][256] f32  ->  wt[256][512] f16 (transposed: contiguous-K for B operand)
__global__ __launch_bounds__(256) void k_cvt_w(const float* __restrict__ W,
                                               _Float16* __restrict__ wt) {
    int i = blockIdx.x * 256 + threadIdx.x;   // i in [0, 512*256)
    int k = i >> 8, n = i & 255;
    wt[n * GAT_INF + k] = (_Float16)W[i];
}

// ----------------------- 2) h = input @ W  (WMMA) --------------------------
// Each wave computes one 16x16 output tile; K = 512 -> 16 x wmma(16x16x32).
// A-operand layout (ISA 7.12.2, 16-bit A 16x32): lane = M (mod 16),
//   elements 0..7  = K[ 8*(lane>=16) + 0..7 ]
//   elements 8..15 = K[16 + 8*(lane>=16) + 0..7 ]
// -> two contiguous 16B loads per lane. B mirrors this with lane = N.

__global__ __launch_bounds__(256) void k_gemm_h(const _Float16* __restrict__ Af,   // input f16 [8192][512]
                                                const _Float16* __restrict__ Bt,   // W^T  f16 [256][512]
                                                float* __restrict__ h,             // h f32 [8192][256]
                                                _Float16* __restrict__ ht) {       // h^T f16 [256][8192]
    const int tid  = threadIdx.x;
    const int wave = tid >> 5, lane = tid & 31;
    const int half = lane >> 4, lm = lane & 15;
    const int mTile = blockIdx.x;                 // 0..511
    const int nTile = blockIdx.y * 8 + wave;      // 0..15

    const _Float16* arow = Af + (size_t)(mTile * 16 + lm) * GAT_INF;
    const _Float16* brow = Bt + (size_t)(nTile * 16 + lm) * GAT_INF;

    v8f acc = {};
    for (int k0 = 0; k0 < GAT_INF; k0 += 32) {
        v16h a = join16(*(const v8h*)(arow + k0 + 8 * half),
                        *(const v8h*)(arow + k0 + 16 + 8 * half));
        v16h b = join16(*(const v8h*)(brow + k0 + 8 * half),
                        *(const v8h*)(brow + k0 + 16 + 8 * half));
        acc = __builtin_amdgcn_wmma_f32_16x16x32_f16(false, a, false, b,
                                                     (short)0, acc, false, false);
    }

    // D layout: lanes 0-15 -> M = r, lanes 16-31 -> M = r + 8, N = lane & 15.
    const int n = nTile * 16 + lm;
    v8h hp;
#pragma unroll
    for (int r = 0; r < 8; ++r) {
        int m = mTile * 16 + r + 8 * half;
        h[(size_t)m * GAT_OUTF + n] = acc[r];
        hp[r] = (_Float16)acc[r];
    }
    // transposed f16 store: 8 consecutive m values -> one 16B store
    *(v8h*)(ht + (size_t)n * GAT_N + mTile * 16 + 8 * half) = hp;
}

// ----------------------- 3) s1 = h@a1, s2 = h@a2 ---------------------------

__global__ __launch_bounds__(256) void k_s12(const float* __restrict__ h,
                                             const float* __restrict__ a,
                                             float* __restrict__ s1,
                                             float* __restrict__ s2) {
    const int wave = threadIdx.x >> 5, lane = threadIdx.x & 31;
    const int row  = blockIdx.x * 8 + wave;
    const float* hr = h + (size_t)row * GAT_OUTF;
    float x1 = 0.f, x2 = 0.f;
#pragma unroll
    for (int n = lane; n < GAT_OUTF; n += 32) {
        float hv = hr[n];
        x1 += hv * a[n];
        x2 += hv * a[GAT_OUTF + n];
    }
#pragma unroll
    for (int off = 16; off > 0; off >>= 1) {
        x1 += __shfl_xor(x1, off, 32);
        x2 += __shfl_xor(x2, off, 32);
    }
    if (lane == 0) { s1[row] = x1; s2[row] = x2; }
}

// --------- 4) row softmax stats + mask bit-pack (single adj pass) ----------
// One block per row i. Wave w handles 32 contiguous 32-wide j-chunks.
// ballot packs adjacency bits 32x (256MB -> 8MB for the second pass).

__global__ __launch_bounds__(256) void k_stats(const int* __restrict__ adj,
                                               const float* __restrict__ s1,
                                               const float* __restrict__ s2,
                                               float* __restrict__ mrow,
                                               float* __restrict__ invl,
                                               unsigned* __restrict__ maskw) {
    const int i = blockIdx.x;
    const int tid = threadIdx.x, wave = tid >> 5, lane = tid & 31;
    const float s1i = s1[i];
    const int* arow = adj + (size_t)i * GAT_N;

    float m = -3.402823466e38f, s = 0.f;
#pragma unroll 4
    for (int t = 0; t < 32; ++t) {
        const int c = wave * 32 + t;          // chunk index 0..255
        const int j = c * 32 + lane;
        if (t + 1 < 32) __builtin_prefetch(arow + j + 32, 0, 1);   // global_prefetch_b8
        const bool bit = arow[j] > 0;
        const unsigned word = gat_ballot32(bit);
        if (lane == 0) maskw[(size_t)i * 256 + c] = word;
        float e = s1i + s2[j];
        e = e > 0.f ? e : GAT_ALPHA * e;
        const float v = bit ? e : GAT_NEGINF;  // finite sentinel, matches reference
        float mn = fmaxf(m, v);
        s = s * __expf(m - mn) + __expf(v - mn);
        m = mn;
    }
#pragma unroll
    for (int off = 16; off > 0; off >>= 1)
        sm_combine(m, s, __shfl_xor(m, off, 32), __shfl_xor(s, off, 32));

    __shared__ float sm8[8], ss8[8];
    if (lane == 0) { sm8[wave] = m; ss8[wave] = s; }
    __syncthreads();
    if (tid == 0) {
        float M = sm8[0], S = ss8[0];
#pragma unroll
        for (int w = 1; w < 8; ++w) sm_combine(M, S, sm8[w], ss8[w]);
        mrow[i] = M;
        invl[i] = 1.0f / S;   // S >= 1 always (max term contributes exp(0))
    }
}

// --------- 5) out = elu(softmax(e) @ h)  (WMMA over probability tiles) -----
// Block = 32 rows (two 16-row M sub-tiles), 8 waves; wave owns 32 columns.
// Per 32-j chunk: 256 threads cooperatively compute the 32x32 f16
// probability tile into LDS (in A-operand order), then each wave loads its
// two B tiles from ht ONCE and uses each in two WMMAs (one per M sub-tile).
// B-reuse across M halves total L2 traffic for ht to ~1 GB.

__global__ __launch_bounds__(256) void k_attn(const _Float16* __restrict__ ht,
                                              const unsigned* __restrict__ maskw,
                                              const float* __restrict__ s1,
                                              const float* __restrict__ s2,
                                              const float* __restrict__ mrow,
                                              const float* __restrict__ invl,
                                              float* __restrict__ out) {
    const int tid = threadIdx.x, wave = tid >> 5, lane = tid & 31;
    const int half = lane >> 4, lm = lane & 15;
    const int row0 = blockIdx.x * 32;

    __shared__ __align__(16) _Float16 sp[32][32];   // probability tile (2 KB)
    __shared__ unsigned smask[32];
    __shared__ float ss1[32], smx[32], sil[32], ss2[32];

    if (tid < 32) {
        int r = row0 + tid;
        ss1[tid] = s1[r]; smx[tid] = mrow[r]; sil[tid] = invl[r];
    }

    // this thread's four probability entries: idx = 4*tid, same row, 4 adjacent j
    const int idx  = tid * 4;
    const int prow = idx >> 5, pj = idx & 31;

    v8f acc00 = {}, acc01 = {}, acc10 = {}, acc11 = {};
    const int nb0 = wave * 32, nb1 = nb0 + 16;
    const _Float16* b0base = ht + (size_t)(nb0 + lm) * GAT_N + 8 * half;
    const _Float16* b1base = ht + (size_t)(nb1 + lm) * GAT_N + 8 * half;

    for (int c = 0; c < 256; ++c) {
        const int j0 = c * 32;
        __syncthreads();   // previous iteration's LDS reads complete
        if (tid < 32)                     smask[tid]    = maskw[(size_t)(row0 + tid) * 256 + c];
        else if (tid >= 32 && tid < 64)   ss2[tid - 32] = s2[j0 + tid - 32];
        __syncthreads();

        {   // p = exp(masked_leakyrelu(s1_i + s2_j) - m_i) * invl_i
            const unsigned w = smask[prow];
            const float m_ = smx[prow], il = sil[prow], s1v = ss1[prow];
#pragma unroll
            for (int q = 0; q < 4; ++q) {
                float e = s1v + ss2[pj + q];
                e = e > 0.f ? e : GAT_ALPHA * e;
                const float v = ((w >> (pj + q)) & 1u) ? e : GAT_NEGINF;
                sp[prow][pj + q] = (_Float16)(__expf(v - m_) * il);
            }
        }
        __syncthreads();

        // A sub-tiles from LDS (ds_load_b128 x2 each); B tiles from ht, each
        // loaded once and consumed by two WMMAs (reuse across M sub-tiles).
        v16h aLo = join16(*(const v8h*)&sp[lm][8 * half],
                          *(const v8h*)&sp[lm][16 + 8 * half]);
        v16h aHi = join16(*(const v8h*)&sp[16 + lm][8 * half],
                          *(const v8h*)&sp[16 + lm][16 + 8 * half]);
        const _Float16* b0p = b0base + j0;
        const _Float16* b1p = b1base + j0;
        v16h b0 = join16(*(const v8h*)b0p, *(const v8h*)(b0p + 16));
        v16h b1 = join16(*(const v8h*)b1p, *(const v8h*)(b1p + 16));

        acc00 = __builtin_amdgcn_wmma_f32_16x16x32_f16(false, aLo, false, b0,
                                                       (short)0, acc00, false, false);
        acc10 = __builtin_amdgcn_wmma_f32_16x16x32_f16(false, aHi, false, b0,
                                                       (short)0, acc10, false, false);
        acc01 = __builtin_amdgcn_wmma_f32_16x16x32_f16(false, aLo, false, b1,
                                                       (short)0, acc01, false, false);
        acc11 = __builtin_amdgcn_wmma_f32_16x16x32_f16(false, aHi, false, b1,
                                                       (short)0, acc11, false, false);

        if (c + 1 < 256) {                       // hide next chunk's B latency
            __builtin_prefetch(b0p + 32, 0, 1);  // global_prefetch_b8
            __builtin_prefetch(b1p + 32, 0, 1);
        }
    }

    // fused ELU + store (D layout: lanes 0-15 -> M=r, 16-31 -> M=r+8)
#pragma unroll
    for (int r = 0; r < 8; ++r) {
        const int mLo = row0 + r + 8 * half;
        const int mHi = mLo + 16;
        float v;
        v = acc00[r]; out[(size_t)mLo * GAT_OUTF + nb0 + lm] = v > 0.f ? v : expm1f(v);
        v = acc01[r]; out[(size_t)mLo * GAT_OUTF + nb1 + lm] = v > 0.f ? v : expm1f(v);
        v = acc10[r]; out[(size_t)mHi * GAT_OUTF + nb0 + lm] = v > 0.f ? v : expm1f(v);
        v = acc11[r]; out[(size_t)mHi * GAT_OUTF + nb1 + lm] = v > 0.f ? v : expm1f(v);
    }
}

// ------------------------------- launch ------------------------------------

extern "C" void kernel_launch(void* const* d_in, const int* in_sizes, int n_in,
                              void* d_out, int out_size, void* d_ws, size_t ws_size,
                              hipStream_t stream) {
    (void)in_sizes; (void)n_in; (void)out_size; (void)ws_size;

    const float* input = (const float*)d_in[0];   // [8192][512]  f32
    const int*   adj   = (const int*)  d_in[1];   // [8192][8192] i32
    const float* W     = (const float*)d_in[2];   // [512][256]   f32
    const float* a     = (const float*)d_in[3];   // [512][1]     f32
    float* out = (float*)d_out;                   // [8192][256]  f32

    // workspace carve-up (all offsets 256B-aligned)
    char* ws = (char*)d_ws;
    size_t off = 0;
    _Float16* input_f16 = (_Float16*)(ws + off); off += (size_t)GAT_N * GAT_INF * 2;      //  8 MB
    _Float16* wt_f16    = (_Float16*)(ws + off); off += (size_t)GAT_OUTF * GAT_INF * 2;   // 256 KB
    _Float16* ht_f16    = (_Float16*)(ws + off); off += (size_t)GAT_OUTF * GAT_N * 2;     //  4 MB
    float*    h_f32     = (float*)   (ws + off); off += (size_t)GAT_N * GAT_OUTF * 4;     //  8 MB
    float*    s1        = (float*)   (ws + off); off += (size_t)GAT_N * 4;
    float*    s2        = (float*)   (ws + off); off += (size_t)GAT_N * 4;
    float*    mrow      = (float*)   (ws + off); off += (size_t)GAT_N * 4;
    float*    invl      = (float*)   (ws + off); off += (size_t)GAT_N * 4;
    unsigned* maskw     = (unsigned*)(ws + off); off += (size_t)GAT_N * 256 * 4;          //  8 MB

    // 1) precision prep
    k_cvt_input<<<(GAT_N * GAT_INF + 255) / 256, 256, 0, stream>>>(input, input_f16,
                                                                   GAT_N * GAT_INF);
    k_cvt_w<<<(GAT_INF * GAT_OUTF + 255) / 256, 256, 0, stream>>>(W, wt_f16);

    // 2) h = input @ W   (WMMA)
    k_gemm_h<<<dim3(GAT_N / 16, 2), 256, 0, stream>>>(input_f16, wt_f16, h_f32, ht_f16);

    // 3) s1 / s2
    k_s12<<<GAT_N / 8, 256, 0, stream>>>(h_f32, a, s1, s2);

    // 4) softmax row stats + 32x mask compression (single pass over adj)
    k_stats<<<GAT_N, 256, 0, stream>>>(adj, s1, s2, mrow, invl, maskw);

    // 5) out = elu(attention @ h)   (WMMA, 32-row M tiles for B reuse)
    k_attn<<<GAT_N / 32, 256, 0, stream>>>(ht_f16, maskw, s1, s2, mrow, invl, out);
}
